// ScaleAdaptiveFusion_52501680226378
// MI455X (gfx1250) — matrix operand order repaired and checked by reading, not compiled, not run on hardware
//
#include <hip/hip_runtime.h>
#include <hip/hip_bf16.h>
#include <math.h>

// Problem constants
constexpr int B  = 16;
constexpr int N  = 16384;
constexpr int C  = 256;
constexpr int S  = 3;
constexpr int BS = B * S;                 // 48
constexpr int ROWS_PER_BLOCK = 256;       // k1: rows of N handled per block
constexpr int CHUNKS = N / ROWS_PER_BLOCK; // 64
constexpr size_t OUT_MAIN = (size_t)B * N * C; // 67,108,864 floats

// Scratch layout (float element offsets into d_ws)
constexpr size_t PS_OFF     = 0;                               // [48][64][256] partial sums
constexpr size_t PM_OFF     = (size_t)BS * CHUNKS * C;         // [48][64][256] partial maxes
constexpr size_t POOLED_OFF = PM_OFF + (size_t)BS * CHUNKS * C; // [48][512] (avg || max)
constexpr size_t WTS_OFF    = POOLED_OFF + (size_t)BS * 2 * C; // [16][3] softmax weights

typedef __attribute__((ext_vector_type(16))) _Float16 v16h;
typedef __attribute__((ext_vector_type(8)))  float    v8f;

__device__ __forceinline__ float wave_sum(float v) {
#pragma unroll
  for (int m = 16; m >= 1; m >>= 1) v += __shfl_xor(v, m, 32);
  return v;
}

// ---------------------------------------------------------------------------
// Kernel 1: per-row LayerNorm stats + normalized sum/max pooling partials.
// grid = (CHUNKS, BS), block = 256 (8 waves, one row per wave per iteration)
// ---------------------------------------------------------------------------
__global__ void __launch_bounds__(256)
k1_pool(const float* __restrict__ f0, const float* __restrict__ f1,
        const float* __restrict__ f2, const float* __restrict__ ln_g,
        const float* __restrict__ ln_b, float* __restrict__ ws) {
  const int chunk = blockIdx.x;        // 0..63
  const int bs    = blockIdx.y;        // 0..47
  const int b     = bs / 3;
  const int s     = bs % 3;
  const float* __restrict__ X = (s == 0) ? f0 : ((s == 1) ? f1 : f2);

  const int lane = threadIdx.x & 31;
  const int wv   = threadIdx.x >> 5;   // 0..7

  // per-lane channels: c = lane*8 + j
  float4 g0 = *(const float4*)(ln_g + s * C + lane * 8);
  float4 g1 = *(const float4*)(ln_g + s * C + lane * 8 + 4);
  float4 t0 = *(const float4*)(ln_b + s * C + lane * 8);
  float4 t1 = *(const float4*)(ln_b + s * C + lane * 8 + 4);
  float g[8] = {g0.x, g0.y, g0.z, g0.w, g1.x, g1.y, g1.z, g1.w};
  float t[8] = {t0.x, t0.y, t0.z, t0.w, t1.x, t1.y, t1.z, t1.w};

  float acs[8], acm[8];
#pragma unroll
  for (int j = 0; j < 8; ++j) { acs[j] = 0.0f; acm[j] = -INFINITY; }

  const size_t base = (size_t)b * N * C;
  const int n0 = chunk * ROWS_PER_BLOCK + wv * 32;

  for (int i = 0; i < 32; ++i) {
    const float* __restrict__ row = X + base + (size_t)(n0 + i) * C;
    // speculative prefetch two rows ahead (global_prefetch_b8)
    __builtin_prefetch(row + 2 * C + lane * 8, 0, 0);
    float4 x0 = *(const float4*)(row + lane * 8);
    float4 x1 = *(const float4*)(row + lane * 8 + 4);
    float x[8] = {x0.x, x0.y, x0.z, x0.w, x1.x, x1.y, x1.z, x1.w};

    float s1 = 0.0f, s2 = 0.0f;
#pragma unroll
    for (int j = 0; j < 8; ++j) { s1 += x[j]; s2 += x[j] * x[j]; }
    s1 = wave_sum(s1);
    s2 = wave_sum(s2);
    const float mu   = s1 * (1.0f / C);
    const float var  = s2 * (1.0f / C) - mu * mu;
    const float rstd = rsqrtf(var + 1e-5f);
#pragma unroll
    for (int j = 0; j < 8; ++j) {
      const float y = (x[j] - mu) * rstd * g[j] + t[j];
      acs[j] += y;
      acm[j] = fmaxf(acm[j], y);
    }
  }

  __shared__ float lsum[8][C];
  __shared__ float lmax[8][C];
#pragma unroll
  for (int j = 0; j < 8; ++j) {
    lsum[wv][lane * 8 + j] = acs[j];
    lmax[wv][lane * 8 + j] = acm[j];
  }
  __syncthreads();

  const int c = threadIdx.x;           // 0..255 (one channel per thread)
  float ts = 0.0f, tm = -INFINITY;
#pragma unroll
  for (int w = 0; w < 8; ++w) { ts += lsum[w][c]; tm = fmaxf(tm, lmax[w][c]); }

  ws[PS_OFF + ((size_t)bs * CHUNKS + chunk) * C + c] = ts;
  ws[PM_OFF + ((size_t)bs * CHUNKS + chunk) * C + c] = tm;
}

// ---------------------------------------------------------------------------
// Kernel 1b: fold partials -> pooled[48][512] = [avg || max]
// grid = 48, block = 256
// ---------------------------------------------------------------------------
__global__ void __launch_bounds__(256)
k1b_reduce(float* __restrict__ ws) {
  const int bs = blockIdx.x;
  const int c  = threadIdx.x;
  float ts = 0.0f, tm = -INFINITY;
  for (int ch = 0; ch < CHUNKS; ++ch) {
    ts += ws[PS_OFF + ((size_t)bs * CHUNKS + ch) * C + c];
    tm = fmaxf(tm, ws[PM_OFF + ((size_t)bs * CHUNKS + ch) * C + c]);
  }
  ws[POOLED_OFF + (size_t)bs * (2 * C) + c]     = ts * (1.0f / N);
  ws[POOLED_OFF + (size_t)bs * (2 * C) + C + c] = tm;
}

// ---------------------------------------------------------------------------
// Kernel 2: MLP (48x512 @ 512x64 via v_wmma_f32_16x16x32_f16) + scores +
// softmax(scores/5). 12 waves, each computes one 16x16 tile of h.
// block = 384
// ---------------------------------------------------------------------------
__global__ void __launch_bounds__(384)
k2_mlp(const float* __restrict__ w1, const float* __restrict__ b1,
       const float* __restrict__ w2, const float* __restrict__ b2,
       float* __restrict__ ws, float* __restrict__ out_w) {
  __shared__ float h[BS][64];
  __shared__ float sc[BS];

  const float* __restrict__ pooled = ws + POOLED_OFF;
  const int tid  = threadIdx.x;
  const int wv   = tid >> 5;           // 0..11
  const int lane = tid & 31;
  const int half = lane >> 4;          // lane half selects K sub-block
  const int mr   = lane & 15;          // M (for A) / N (for B,D) index
  const int mt   = wv >> 2;            // 0..2 : M tile (rows of pooled)
  const int nt   = wv & 3;             // 0..3 : N tile (cols of w1)

  const float* __restrict__ prow = pooled + (size_t)(mt * 16 + mr) * (2 * C);

  v8f acc = {};
  for (int kc = 0; kc < 16; ++kc) {    // K = 512 in steps of 32
    v16h a, bm;
#pragma unroll
    for (int v = 0; v < 8; ++v) {
#pragma unroll
      for (int p = 0; p < 2; ++p) {
        // A (16x32 f16): lanes 0-15 hold K 0..7 (v0..3) & 16..23 (v4..7);
        // lanes 16-31 hold K 8..15 & 24..31.
        const int Ka = (v < 4) ? (8 * half + 2 * v + p)
                               : (16 + 8 * half + 2 * (v - 4) + p);
        a[2 * v + p] = (_Float16)prow[kc * 32 + Ka];
        // B (32x16 f16): same K order as A (lanes 0-15 K 0..7 & 16..23; lanes 16-31 K 8..15 & 24..31).
        const int Kb = Ka;
        bm[2 * v + p] = (_Float16)w1[(size_t)(kc * 32 + Kb) * 64 + nt * 16 + mr];
      }
    }
    acc = __builtin_amdgcn_wmma_f32_16x16x32_f16(
        false, a, false, bm, (short)0, acc, false, false);
  }
  // D layout: VGPR j -> row M = j + 8*half, col N = mr
#pragma unroll
  for (int j = 0; j < 8; ++j) {
    const int row = mt * 16 + j + 8 * half;
    const int col = nt * 16 + mr;
    h[row][col] = fmaxf(acc[j] + b1[col], 0.0f);
  }
  __syncthreads();

  if (tid < BS) {
    float a2 = b2[0];
#pragma unroll
    for (int f = 0; f < 64; ++f) a2 += h[tid][f] * w2[f];
    sc[tid] = a2;
  }
  __syncthreads();

  if (tid < B) {
    const int bb = tid;
    const float s0 = sc[bb * 3 + 0] * 0.2f;
    const float s1 = sc[bb * 3 + 1] * 0.2f;
    const float s2 = sc[bb * 3 + 2] * 0.2f;
    const float m  = fmaxf(s0, fmaxf(s1, s2));
    const float e0 = __expf(s0 - m), e1 = __expf(s1 - m), e2 = __expf(s2 - m);
    const float inv = 1.0f / (e0 + e1 + e2);
    float* wts = ws + WTS_OFF;
    wts[bb * 3 + 0] = e0 * inv;  out_w[bb * 3 + 0] = e0 * inv;
    wts[bb * 3 + 1] = e1 * inv;  out_w[bb * 3 + 1] = e1 * inv;
    wts[bb * 3 + 2] = e2 * inv;  out_w[bb * 3 + 2] = e2 * inv;
  }
}

// ---------------------------------------------------------------------------
// Kernel 3: weighted fusion of (recomputed) normalized scales + final LN.
// grid = B*N/8, block = 256 (one row per wave)
// ---------------------------------------------------------------------------
__global__ void __launch_bounds__(256)
k3_fuse(const float* __restrict__ f0, const float* __restrict__ f1,
        const float* __restrict__ f2, const float* __restrict__ ln_g,
        const float* __restrict__ ln_b, const float* __restrict__ fg,
        const float* __restrict__ fb, const float* __restrict__ ws,
        float* __restrict__ out) {
  const float* __restrict__ wts = ws + WTS_OFF;
  const int lane = threadIdx.x & 31;
  const int wv   = threadIdx.x >> 5;
  const size_t r   = (size_t)blockIdx.x * 8 + wv;   // row over B*N
  const int    b   = (int)(r >> 14);                // / 16384
  const size_t off = r * C + lane * 8;

  // warm the pipe for scales 1 and 2 while scale 0 loads
  __builtin_prefetch(f1 + off, 0, 0);
  __builtin_prefetch(f2 + off, 0, 0);

  float fuse[8] = {0, 0, 0, 0, 0, 0, 0, 0};
#pragma unroll
  for (int s = 0; s < 3; ++s) {
    const float* __restrict__ X = (s == 0) ? f0 : ((s == 1) ? f1 : f2);
    float4 x0 = *(const float4*)(X + off);
    float4 x1 = *(const float4*)(X + off + 4);
    float x[8] = {x0.x, x0.y, x0.z, x0.w, x1.x, x1.y, x1.z, x1.w};

    float s1 = 0.0f, s2 = 0.0f;
#pragma unroll
    for (int j = 0; j < 8; ++j) { s1 += x[j]; s2 += x[j] * x[j]; }
    s1 = wave_sum(s1);
    s2 = wave_sum(s2);
    const float mu   = s1 * (1.0f / C);
    const float rstd = rsqrtf(s2 * (1.0f / C) - mu * mu + 1e-5f);
    const float w    = wts[b * 3 + s];

    float4 g0 = *(const float4*)(ln_g + s * C + lane * 8);
    float4 g1 = *(const float4*)(ln_g + s * C + lane * 8 + 4);
    float4 t0 = *(const float4*)(ln_b + s * C + lane * 8);
    float4 t1 = *(const float4*)(ln_b + s * C + lane * 8 + 4);
    float g[8] = {g0.x, g0.y, g0.z, g0.w, g1.x, g1.y, g1.z, g1.w};
    float t[8] = {t0.x, t0.y, t0.z, t0.w, t1.x, t1.y, t1.z, t1.w};
#pragma unroll
    for (int j = 0; j < 8; ++j)
      fuse[j] += w * ((x[j] - mu) * rstd * g[j] + t[j]);
  }

  // final LayerNorm over fused row
  float s1 = 0.0f, s2 = 0.0f;
#pragma unroll
  for (int j = 0; j < 8; ++j) { s1 += fuse[j]; s2 += fuse[j] * fuse[j]; }
  s1 = wave_sum(s1);
  s2 = wave_sum(s2);
  const float mu   = s1 * (1.0f / C);
  const float rstd = rsqrtf(s2 * (1.0f / C) - mu * mu + 1e-5f);

  float4 G0 = *(const float4*)(fg + lane * 8);
  float4 G1 = *(const float4*)(fg + lane * 8 + 4);
  float4 T0 = *(const float4*)(fb + lane * 8);
  float4 T1 = *(const float4*)(fb + lane * 8 + 4);
  float G[8] = {G0.x, G0.y, G0.z, G0.w, G1.x, G1.y, G1.z, G1.w};
  float T[8] = {T0.x, T0.y, T0.z, T0.w, T1.x, T1.y, T1.z, T1.w};

  float4 o0, o1;
  o0.x = (fuse[0] - mu) * rstd * G[0] + T[0];
  o0.y = (fuse[1] - mu) * rstd * G[1] + T[1];
  o0.z = (fuse[2] - mu) * rstd * G[2] + T[2];
  o0.w = (fuse[3] - mu) * rstd * G[3] + T[3];
  o1.x = (fuse[4] - mu) * rstd * G[4] + T[4];
  o1.y = (fuse[5] - mu) * rstd * G[5] + T[5];
  o1.z = (fuse[6] - mu) * rstd * G[6] + T[6];
  o1.w = (fuse[7] - mu) * rstd * G[7] + T[7];
  *(float4*)(out + off)     = o0;
  *(float4*)(out + off + 4) = o1;
}

// ---------------------------------------------------------------------------
extern "C" void kernel_launch(void* const* d_in, const int* in_sizes, int n_in,
                              void* d_out, int out_size, void* d_ws, size_t ws_size,
                              hipStream_t stream) {
  const float* f0   = (const float*)d_in[0];
  const float* f1   = (const float*)d_in[1];
  const float* f2   = (const float*)d_in[2];
  const float* ln_g = (const float*)d_in[3];
  const float* ln_b = (const float*)d_in[4];
  const float* w1   = (const float*)d_in[5];
  const float* b1   = (const float*)d_in[6];
  const float* w2   = (const float*)d_in[7];
  const float* b2   = (const float*)d_in[8];
  const float* fg   = (const float*)d_in[9];
  const float* fb   = (const float*)d_in[10];

  float* ws  = (float*)d_ws;
  float* out = (float*)d_out;

  // Pass 1: pooled partials (streams 768 MB)
  k1_pool<<<dim3(CHUNKS, BS), 256, 0, stream>>>(f0, f1, f2, ln_g, ln_b, ws);
  // Fold partials -> pooled[48][512]
  k1b_reduce<<<BS, 256, 0, stream>>>(ws);
  // Tiny WMMA MLP + softmax -> weights (scratch + output tail)
  k2_mlp<<<1, 384, 0, stream>>>(w1, b1, w2, b2, ws, out + OUT_MAIN);
  // Pass 2: fusion + final LN (streams 768 MB in, 256 MB out)
  k3_fuse<<<(B * N) / 8, 256, 0, stream>>>(f0, f1, f2, ln_g, ln_b, fg, fb, ws, out);
}
